// ReactNetBlock_75977971466568
// MI455X (gfx1250) — matrix operand order repaired and checked by reading, not compiled, not run on hardware
//
#include <hip/hip_runtime.h>
#include <hip/hip_bf16.h>
#include <math.h>

// ---------------------------------------------------------------------------
// ReactNetBlock fused pipeline for gfx1250 (MI455X, wave32, WMMA).
// ~270 GFLOP of GEMM vs ~130MB traffic -> matrix-op bound -> f16 WMMA path.
// GEMM: 128x128 block tile, 32x64 wave tile, K-step 64, double-buffered
// async global->LDS staging (ASYNCcnt), fused bias/SwiGLU/SiLU/combine
// epilogues staged through LDS.
// ---------------------------------------------------------------------------

typedef __attribute__((ext_vector_type(16))) _Float16 v16h;
typedef __attribute__((ext_vector_type(8)))  _Float16 v8h;
typedef __attribute__((ext_vector_type(8)))  float    v8f;

#define DIMC   384
#define INNERC 1536
#define NEXP   5
#define KSZ    31
#define SQC    48
#define BC     4
#define TC     1024
#define NTOK   (BC * TC)

#define BT_M 128
#define BT_N 128
#define BT_K 64   // K-depth per staged tile (two 32-deep WMMA sub-steps)

__device__ __forceinline__ float silu_f(float x) {
    return x / (1.0f + __expf(-x));
}

// CDNA5 async global->LDS copy (ASYNCcnt), no VGPR round trip.
__device__ __forceinline__ void async_load_b128(unsigned lds_off, const void* gptr) {
    asm volatile("global_load_async_to_lds_b128 %0, %1, off"
                 :: "v"(lds_off), "v"((unsigned long long)(uintptr_t)gptr)
                 : "memory");
}
__device__ __forceinline__ void wait_asynccnt0() {
    asm volatile("s_wait_asynccnt 0x0" ::: "memory");
}

// ---------------------------------------------------------------------------
// Weight convert: fp32 [E][K][N] row-major  ->  f16 [E][N'][K] (B^T layout for
// K-contiguous LDS loads). interleave=1 maps n' -> orig col (a/g interleave)
// so SwiGLU pairs (2j, 2j+1) are adjacent output columns.
// ---------------------------------------------------------------------------
__global__ void convert_w_kernel(const float* __restrict__ src,
                                 _Float16* __restrict__ dst,
                                 int K, int N, int interleave, size_t total) {
    size_t o = (size_t)blockIdx.x * 256 + threadIdx.x;
    if (o >= total) return;
    size_t kn = (size_t)K * N;
    int e  = (int)(o / kn);
    int r  = (int)(o % kn);
    int np = r / K;
    int k  = r % K;
    int j  = interleave ? ((np & 1) ? (N >> 1) + (np >> 1) : (np >> 1)) : np;
    dst[o] = (_Float16)src[(size_t)e * kn + (size_t)k * N + j];
}

// ---------------------------------------------------------------------------
// LayerNorm (block per token) + s = xln * senet_res
// ---------------------------------------------------------------------------
__global__ void ln_kernel(const float* __restrict__ res,
                          const float* __restrict__ g,
                          const float* __restrict__ b,
                          const float* __restrict__ sres,
                          float* __restrict__ xln,
                          float* __restrict__ s) {
    __shared__ float red[256];
    const int n = blockIdx.x;
    const float* x = res + (size_t)n * DIMC;
    const int tid = threadIdx.x;

    float part = 0.f;
    for (int d = tid; d < DIMC; d += 256) part += x[d];
    red[tid] = part; __syncthreads();
    for (int st = 128; st > 0; st >>= 1) {
        if (tid < st) red[tid] += red[tid + st];
        __syncthreads();
    }
    const float mu = red[0] / (float)DIMC;
    __syncthreads();

    part = 0.f;
    for (int d = tid; d < DIMC; d += 256) { float t = x[d] - mu; part += t * t; }
    red[tid] = part; __syncthreads();
    for (int st = 128; st > 0; st >>= 1) {
        if (tid < st) red[tid] += red[tid + st];
        __syncthreads();
    }
    const float rstd = rsqrtf(red[0] / (float)DIMC + 1e-5f);

    for (int d = tid; d < DIMC; d += 256) {
        float v = (x[d] - mu) * rstd * g[d] + b[d];
        xln[(size_t)n * DIMC + d] = v;
        s[(size_t)n * DIMC + d]   = v * sres[d];
    }
}

// ---------------------------------------------------------------------------
// Depthwise conv over time (pad 15): writes h f32 (y accumulator init) + f16.
// ---------------------------------------------------------------------------
__global__ void conv_kernel(const float* __restrict__ xln,
                            const float* __restrict__ cw,
                            const float* __restrict__ cb,
                            float* __restrict__ y,
                            _Float16* __restrict__ tokh) {
    int idx = blockIdx.x * 256 + threadIdx.x;
    if (idx >= NTOK * DIMC) return;
    const int d  = idx % DIMC;
    const int n  = idx / DIMC;
    const int t  = n % TC;
    const int bb = n / TC;
    float acc = cb[d];
#pragma unroll 1
    for (int k = 0; k < KSZ; ++k) {
        int tt = t + k - (KSZ / 2);
        if (tt >= 0 && tt < TC)
            acc += xln[((size_t)bb * TC + tt) * DIMC + d] * cw[d * KSZ + k];
    }
    y[idx]    = acc;
    tokh[idx] = (_Float16)acc;
}

// ---------------------------------------------------------------------------
// WMMA GEMM: C[M,N] = A[M,K](f16 row-major) x B^T[N,K](f16) (+bias, +act).
// 256 threads = 8 waves; block tile 128x128; wave tile 32x64 (2x4 frags of
// v_wmma_f32_16x16x32_f16); K-step 64, DOUBLE-BUFFERED async global->LDS:
// tile k+1 is issued before computing tile k, so s_wait_asynccnt at the top
// of the next iteration finds data already landed. Cs (f32 epilogue staging,
// 64KB) aliases both A/B buffers (dead after K loop).
// Epilogue modes:
//   0: SwiGLU on interleaved column pairs -> f16 out [M, N/2]
//   1: SiLU -> f16 out [M, N]
//   2: out_f32[m, n] += wvec[m*5+eidx] * (C + bias)   (expert combine)
// ---------------------------------------------------------------------------
__launch_bounds__(256)
__global__ void gemm_wmma_kernel(const _Float16* __restrict__ A,
                                 const _Float16* __restrict__ Bt,
                                 int K, int mode,
                                 const float* __restrict__ biasA,
                                 const float* __restrict__ biasG,
                                 void* __restrict__ outp, int outld,
                                 const float* __restrict__ wvec, int eidx) {
    __shared__ __align__(16) unsigned char smem[BT_M * BT_N * 4]; // 64KB
    _Float16* smh = (_Float16*)smem;
    // buffer b: A at smh + b*16384, B at smh + b*16384 + 8192  (halfs)
    float* Cs = (float*)smem;            // [128][128] f32, aliases buffers

    const int tid   = threadIdx.x;
    const int lane  = tid & 31;
    const int wid   = tid >> 5;
    const int wm    = (wid >> 1) * 32;   // wave row offset (4 rows of waves)
    const int wn    = (wid & 1) * 64;    // wave col offset (2 cols of waves)
    const int mbase = blockIdx.x * BT_M;
    const int nbase = blockIdx.y * BT_N;
    const int l16   = lane & 15;
    const int hi    = lane >> 4;         // 0 | 1

    // cooperative staging: 128 rows x 64 halfs per matrix = 1024 16B chunks;
    // thread covers rows (tid>>3)+32i, half-col (tid&7)*8, i = 0..3.
    const int crow = tid >> 3;
    const int ckc  = (tid & 7) * 8;

    auto stage = [&](int k0, int buf) {
        _Float16* Ab = smh + buf * 16384;
        _Float16* Bb = Ab + 8192;
#pragma unroll
        for (int i = 0; i < 4; ++i) {
            int row = crow + 32 * i;
            async_load_b128((unsigned)(uintptr_t)&Ab[row * BT_K + ckc],
                            &A[(size_t)(mbase + row) * K + k0 + ckc]);
            async_load_b128((unsigned)(uintptr_t)&Bb[row * BT_K + ckc],
                            &Bt[(size_t)(nbase + row) * K + k0 + ckc]);
        }
    };

    v8f acc[2][4] = {};

    stage(0, 0);                          // prologue: tile 0 -> buffer 0
    int buf = 0;
    for (int k0 = 0; k0 < K; k0 += BT_K) {
        wait_asynccnt0();                 // tile k0 landed in `buf`
        __syncthreads();
        if (k0 + BT_K < K) stage(k0 + BT_K, buf ^ 1);   // overlap next tile
        if (k0 + 2 * BT_K < K) {
            __builtin_prefetch(&A[(size_t)(mbase + crow) * K + k0 + 2 * BT_K + ckc], 0, 1);
            __builtin_prefetch(&Bt[(size_t)(nbase + crow) * K + k0 + 2 * BT_K + ckc], 0, 1);
        }

        const _Float16* Ab = smh + buf * 16384;
        const _Float16* Bb = Ab + 8192;
#pragma unroll
        for (int kk = 0; kk < BT_K; kk += 32) {
            // --- fragment loads from LDS (2x ds_load_b128 per fragment) ---
            v16h afr[2], bfr[4];
#pragma unroll
            for (int i = 0; i < 2; ++i) {
                int m  = wm + i * 16 + l16;
                int kb = kk + hi * 8;
                v8h lo = *(const v8h*)&Ab[m * BT_K + kb];
                v8h hh = *(const v8h*)&Ab[m * BT_K + 16 + kb];
#pragma unroll
                for (int j = 0; j < 8; ++j) { afr[i][j] = lo[j]; afr[i][8 + j] = hh[j]; }
            }
#pragma unroll
            for (int i = 0; i < 4; ++i) {
                int n  = wn + i * 16 + l16;
                int kh = kk + hi * 8;
                v8h lo = *(const v8h*)&Bb[n * BT_K + kh];
                v8h hh = *(const v8h*)&Bb[n * BT_K + kh + 16];
#pragma unroll
                for (int j = 0; j < 8; ++j) { bfr[i][j] = lo[j]; bfr[i][8 + j] = hh[j]; }
            }
            // --- 8 WMMAs per sub-step (32x64 wave tile) ---
#pragma unroll
            for (int i = 0; i < 2; ++i)
#pragma unroll
                for (int j = 0; j < 4; ++j)
                    acc[i][j] = __builtin_amdgcn_wmma_f32_16x16x32_f16(
                        false, afr[i], false, bfr[j], (short)0, acc[i][j],
                        false, false);
        }
        buf ^= 1;
    }

    // --- spill accumulators to Cs (C layout: VGPR r -> M=r / M=8+r) ---
    __syncthreads();       // all waves done reading buffers (Cs aliases them)
#pragma unroll
    for (int i = 0; i < 2; ++i)
#pragma unroll
        for (int j = 0; j < 4; ++j)
#pragma unroll
            for (int r = 0; r < 8; ++r) {
                int m = wm + i * 16 + r + hi * 8;
                int n = wn + j * 16 + l16;
                Cs[m * BT_N + n] = acc[i][j][r];
            }
    __syncthreads();

    // --- fused epilogue ---
    if (mode == 0) {                    // SwiGLU: pairs (2j, 2j+1) -> 1 col
        _Float16* out = (_Float16*)outp;
        const int nc2 = nbase >> 1;
#pragma unroll 4
        for (int i = 0; i < 32; ++i) {
            int idx = tid + i * 256;    // 128 x 64 outputs
            int m = idx >> 6, j = idx & 63;
            float a = Cs[m * BT_N + 2 * j]     + biasA[nc2 + j];
            float g = Cs[m * BT_N + 2 * j + 1] + biasG[nc2 + j];
            out[(size_t)(mbase + m) * outld + nc2 + j] = (_Float16)(a * silu_f(g));
        }
    } else if (mode == 1) {             // SiLU (router hidden)
        _Float16* out = (_Float16*)outp;
#pragma unroll 4
        for (int i = 0; i < 64; ++i) {
            int idx = tid + i * 256;    // 128 x 128 outputs
            int m = idx >> 7, j = idx & 127;
            float v = Cs[m * BT_N + j] + biasA[nbase + j];
            out[(size_t)(mbase + m) * outld + nbase + j] = (_Float16)silu_f(v);
        }
    } else {                            // expert combine: y += w * (C + bias)
        float* out = (float*)outp;
#pragma unroll 4
        for (int i = 0; i < 64; ++i) {
            int idx = tid + i * 256;
            int m = idx >> 7, j = idx & 127;
            float v  = Cs[m * BT_N + j] + biasA[nbase + j];
            float ww = wvec[(size_t)(mbase + m) * NEXP + eidx];
            out[(size_t)(mbase + m) * outld + nbase + j] += ww * v;
        }
    }
}

// ---------------------------------------------------------------------------
// Router tail: rv = act_r @ rw2 + rb2 ; top-2 ; masked softmax -> w[N,5]
// ---------------------------------------------------------------------------
__global__ void router_kernel(const _Float16* __restrict__ actr,
                              const float* __restrict__ rw2,
                              const float* __restrict__ rb2,
                              float* __restrict__ w) {
    int n = blockIdx.x * 64 + threadIdx.x;
    if (n >= NTOK) return;
    float rv[NEXP];
#pragma unroll
    for (int e = 0; e < NEXP; ++e) rv[e] = rb2[e];
    for (int d = 0; d < DIMC; ++d) {
        float a = (float)actr[(size_t)n * DIMC + d];
#pragma unroll
        for (int e = 0; e < NEXP; ++e) rv[e] += a * rw2[d * NEXP + e];
    }
    int i0 = 0;
#pragma unroll
    for (int e = 1; e < NEXP; ++e) if (rv[e] > rv[i0]) i0 = e;
    int i1 = (i0 == 0) ? 1 : 0;
#pragma unroll
    for (int e = 0; e < NEXP; ++e)
        if (e != i0 && rv[e] > rv[i1]) i1 = e;
    float mx  = fmaxf(rv[i0], rv[i1]);
    float e0  = __expf(rv[i0] - mx), e1 = __expf(rv[i1] - mx);
    float inv = 1.f / (e0 + e1);
#pragma unroll
    for (int e = 0; e < NEXP; ++e) w[(size_t)n * NEXP + e] = 0.f;
    w[(size_t)n * NEXP + i0] = e0 * inv;
    w[(size_t)n * NEXP + i1] = e1 * inv;
}

// ---------------------------------------------------------------------------
// SE pooling: pooled[b,d] = mean_t (s + y)
// ---------------------------------------------------------------------------
__global__ void pooled_kernel(const float* __restrict__ s,
                              const float* __restrict__ y,
                              float* __restrict__ pooled) {
    int idx = blockIdx.x * 256 + threadIdx.x;
    if (idx >= BC * DIMC) return;
    int b = idx / DIMC, d = idx % DIMC;
    float acc = 0.f;
    for (int t = 0; t < TC; ++t) {
        size_t n = (size_t)b * TC + t;
        acc += s[n * DIMC + d] + y[n * DIMC + d];
    }
    pooled[idx] = acc / (float)TC;
}

// ---------------------------------------------------------------------------
// gate = sigmoid(silu(pooled @ sw1 + sb1) @ sw2 + sb2)   (single block)
// ---------------------------------------------------------------------------
__global__ void gate_kernel(const float* __restrict__ pooled,
                            const float* __restrict__ sw1,
                            const float* __restrict__ sb1,
                            const float* __restrict__ sw2,
                            const float* __restrict__ sb2,
                            float* __restrict__ gate) {
    __shared__ float u[BC * SQC];
    int tid = threadIdx.x;
    if (tid < BC * SQC) {
        int b = tid / SQC, q = tid % SQC;
        float acc = sb1[q];
        for (int d = 0; d < DIMC; ++d) acc += pooled[b * DIMC + d] * sw1[d * SQC + q];
        u[tid] = silu_f(acc);
    }
    __syncthreads();
    if (tid < DIMC) {
        for (int b = 0; b < BC; ++b) {
            float acc = sb2[tid];
            for (int q = 0; q < SQC; ++q) acc += u[b * SQC + q] * sw2[q * DIMC + tid];
            gate[b * DIMC + tid] = 1.f / (1.f + __expf(-acc));
        }
    }
}

// ---------------------------------------------------------------------------
// out = res + y * gate
// ---------------------------------------------------------------------------
__global__ void final_kernel(const float* __restrict__ res,
                             const float* __restrict__ y,
                             const float* __restrict__ gate,
                             float* __restrict__ out) {
    int idx = blockIdx.x * 256 + threadIdx.x;
    if (idx >= NTOK * DIMC) return;
    int d = idx % DIMC;
    int b = idx / (TC * DIMC);
    out[idx] = res[idx] + y[idx] * gate[b * DIMC + d];
}

// ---------------------------------------------------------------------------
extern "C" void kernel_launch(void* const* d_in, const int* in_sizes, int n_in,
                              void* d_out, int out_size, void* d_ws, size_t ws_size,
                              hipStream_t stream) {
    (void)in_sizes; (void)n_in; (void)out_size; (void)ws_size;

    const float* res   = (const float*)d_in[0];
    const float* ln_g  = (const float*)d_in[1];
    const float* ln_b  = (const float*)d_in[2];
    const float* convw = (const float*)d_in[3];
    const float* convb = (const float*)d_in[4];
    const float* rw1   = (const float*)d_in[5];
    const float* rb1   = (const float*)d_in[6];
    const float* rw2   = (const float*)d_in[7];
    const float* rb2   = (const float*)d_in[8];
    const float* we1   = (const float*)d_in[9];
    const float* be1   = (const float*)d_in[10];
    const float* we2   = (const float*)d_in[11];
    const float* be2   = (const float*)d_in[12];
    const float* we3   = (const float*)d_in[13];
    const float* be3   = (const float*)d_in[14];
    const float* sw1   = (const float*)d_in[15];
    const float* sb1   = (const float*)d_in[16];
    const float* sw2   = (const float*)d_in[17];
    const float* sb2   = (const float*)d_in[18];
    const float* sres  = (const float*)d_in[19];
    float* out = (float*)d_out;

    // ---- workspace carve-up (256B aligned) ----
    size_t off = 0;
    char* base = (char*)d_ws;
    auto alloc = [&](size_t bytes) {
        void* p = base + off;
        off += (bytes + 255) & ~(size_t)255;
        return p;
    };
    float*    xln    = (float*)   alloc((size_t)NTOK * DIMC * 4);
    float*    s_buf  = (float*)   alloc((size_t)NTOK * DIMC * 4);
    float*    y_buf  = (float*)   alloc((size_t)NTOK * DIMC * 4);
    _Float16* tokh   = (_Float16*)alloc((size_t)NTOK * DIMC * 2);
    _Float16* actr   = (_Float16*)alloc((size_t)NTOK * DIMC * 2);
    float*    wbuf   = (float*)   alloc((size_t)NTOK * NEXP * 4);
    _Float16* rw1t   = (_Float16*)alloc((size_t)DIMC * DIMC * 2);
    _Float16* we1t   = (_Float16*)alloc((size_t)NEXP * 2 * INNERC * DIMC * 2);
    _Float16* we2t   = (_Float16*)alloc((size_t)NEXP * 2 * INNERC * INNERC * 2);
    _Float16* we3t   = (_Float16*)alloc((size_t)NEXP * DIMC * INNERC * 2);
    _Float16* h1     = (_Float16*)alloc((size_t)NTOK * INNERC * 2);
    _Float16* h2     = (_Float16*)alloc((size_t)NTOK * INNERC * 2);
    float*    pooled = (float*)   alloc((size_t)BC * DIMC * 4);
    float*    gate   = (float*)   alloc((size_t)BC * DIMC * 4);

    // ---- weight conversion to f16 B^T layouts ----
    {
        size_t t;
        t = (size_t)DIMC * DIMC;
        convert_w_kernel<<<(unsigned)((t + 255) / 256), 256, 0, stream>>>(
            rw1, rw1t, DIMC, DIMC, 0, t);
        t = (size_t)NEXP * DIMC * 2 * INNERC;
        convert_w_kernel<<<(unsigned)((t + 255) / 256), 256, 0, stream>>>(
            we1, we1t, DIMC, 2 * INNERC, 1, t);
        t = (size_t)NEXP * INNERC * 2 * INNERC;
        convert_w_kernel<<<(unsigned)((t + 255) / 256), 256, 0, stream>>>(
            we2, we2t, INNERC, 2 * INNERC, 1, t);
        t = (size_t)NEXP * INNERC * DIMC;
        convert_w_kernel<<<(unsigned)((t + 255) / 256), 256, 0, stream>>>(
            we3, we3t, INNERC, DIMC, 0, t);
    }

    // ---- LayerNorm + depthwise conv ----
    ln_kernel<<<NTOK, 256, 0, stream>>>(res, ln_g, ln_b, sres, xln, s_buf);
    conv_kernel<<<(NTOK * DIMC + 255) / 256, 256, 0, stream>>>(
        xln, convw, convb, y_buf, tokh);

    // ---- router: GEMM (SiLU epilogue) + top-2 softmax ----
    gemm_wmma_kernel<<<dim3(NTOK / BT_M, DIMC / BT_N), 256, 0, stream>>>(
        tokh, rw1t, DIMC, /*mode=*/1, rb1, nullptr, actr, DIMC, nullptr, 0);
    router_kernel<<<NTOK / 64, 64, 0, stream>>>(actr, rw2, rb2, wbuf);

    // ---- experts (dense-equivalent): 3 chained WMMA GEMMs each ----
    for (int e = 0; e < NEXP; ++e) {
        const _Float16* w1 = we1t + (size_t)e * 2 * INNERC * DIMC;
        const _Float16* w2 = we2t + (size_t)e * 2 * INNERC * INNERC;
        const _Float16* w3 = we3t + (size_t)e * DIMC * INNERC;
        const float* b1 = be1 + (size_t)e * 2 * INNERC;
        const float* b2 = be2 + (size_t)e * 2 * INNERC;
        const float* b3 = be3 + (size_t)e * DIMC;

        // h1 = swiglu(tok @ we1 + be1)       [N, 1536] f16
        gemm_wmma_kernel<<<dim3(NTOK / BT_M, (2 * INNERC) / BT_N), 256, 0, stream>>>(
            tokh, w1, DIMC, /*mode=*/0, b1, b1 + INNERC, h1, INNERC, nullptr, 0);
        // h2 = swiglu(h1 @ we2 + be2)        [N, 1536] f16
        gemm_wmma_kernel<<<dim3(NTOK / BT_M, (2 * INNERC) / BT_N), 256, 0, stream>>>(
            h1, w2, INNERC, /*mode=*/0, b2, b2 + INNERC, h2, INNERC, nullptr, 0);
        // y += w[:,e] * (h2 @ we3 + be3)     [N, 384] f32
        gemm_wmma_kernel<<<dim3(NTOK / BT_M, DIMC / BT_N), 256, 0, stream>>>(
            h2, w3, INNERC, /*mode=*/2, b3, nullptr, y_buf, DIMC, wbuf, e);
    }

    // ---- SE gate + residual ----
    pooled_kernel<<<(BC * DIMC + 255) / 256, 256, 0, stream>>>(s_buf, y_buf, pooled);
    gate_kernel<<<1, DIMC, 0, stream>>>(pooled, sw1, sb1, sw2, sb2, gate);
    final_kernel<<<(NTOK * DIMC + 255) / 256, 256, 0, stream>>>(res, y_buf, gate, out);
}
